// PathConFFN_39041252720861
// MI455X (gfx1250) — matrix operand order repaired and checked by reading, not compiled, not run on hardware
//
#include <hip/hip_runtime.h>
#include <hip/hip_bf16.h>
#include <stdint.h>

#define NN 50000
#define EE 400000
#define DIN 128
#define DOUT 128
#define DNODE 6
#define NRPAD 160   // node_rep bf16 padded width: 134 -> 160 (5 chunks of 32)

typedef __attribute__((ext_vector_type(16))) __bf16 v16bf;
typedef __attribute__((ext_vector_type(8)))  __bf16 bf16x8;
typedef __attribute__((ext_vector_type(8)))  float  v8f;
typedef __attribute__((ext_vector_type(4)))  float  f32x4;

static __device__ __forceinline__ __bf16 f2bf(float f) {
  unsigned u = __builtin_bit_cast(unsigned, f);
  unsigned r = (u + 0x7FFFu + ((u >> 16) & 1u)) >> 16;
  unsigned short s = (unsigned short)r;
  return __builtin_bit_cast(__bf16, s);
}

#define WMMA_BF16(A, B, C) \
  __builtin_amdgcn_wmma_f32_16x16x32_bf16(false, (A), false, (B), (short)0, (C), false, false)

// A fragment (16x32 bf16, lane = l): row = l&15, kbase = (l>>4)*8:
// elements 0..7 = K kb..kb+7, elements 8..15 = K kb+16..kb+23.
static __device__ __forceinline__ v16bf loadA_bf(const __bf16* rowp, int kb) {
  bf16x8 a0 = *(const bf16x8*)(rowp + kb);
  bf16x8 a1 = *(const bf16x8*)(rowp + kb + 16);
  return __builtin_shufflevector(a0, a1, 0,1,2,3,4,5,6,7,8,9,10,11,12,13,14,15);
}

static __device__ __forceinline__ v16bf loadA_f32(const float* rowp, int kb) {
  f32x4 f0 = *(const f32x4*)(rowp + kb);
  f32x4 f1 = *(const f32x4*)(rowp + kb + 4);
  f32x4 f2 = *(const f32x4*)(rowp + kb + 16);
  f32x4 f3 = *(const f32x4*)(rowp + kb + 20);
  v16bf r;
#pragma unroll
  for (int i = 0; i < 4; ++i) {
    r[i]      = f2bf(f0[i]);
    r[4 + i]  = f2bf(f1[i]);
    r[8 + i]  = f2bf(f2[i]);
    r[12 + i] = f2bf(f3[i]);
  }
  return r;
}

static __device__ __forceinline__ v16bf loadA_lds_f32(const float* rowp, int kb) {
  v16bf r;
#pragma unroll
  for (int i = 0; i < 8; ++i) {
    r[i]     = f2bf(rowp[kb + i]);
    r[8 + i] = f2bf(rowp[kb + 16 + i]);
  }
  return r;
}

// ---------------------------------------------------------------- helpers ---
__global__ void zero_kernel(float* p, int n) {
  int i = blockIdx.x * blockDim.x + threadIdx.x;
  if (i < n) p[i] = 0.f;
}

// 32 threads per edge: 4 columns each; lane 0 also accumulates the denominator.
__global__ void scatter_kernel(const int* __restrict__ ei, const float* __restrict__ ea,
                               const float* __restrict__ mask,
                               float* __restrict__ agg, float* __restrict__ den) {
  int idx = blockIdx.x * blockDim.x + threadIdx.x;
  int e = idx >> 5;
  int q = idx & 31;
  if (e >= EE) return;
  float m = mask[e];
  int c = ei[EE + e];
  if (m != 0.f) {
    if (q == 0) atomicAdd(&den[c], m);
    const float* src = ea + (size_t)e * DIN + q * 4;
    float* dst = agg + (size_t)c * DIN + q * 4;
#pragma unroll
    for (int j = 0; j < 4; ++j) atomicAdd(&dst[j], src[j] * m);
  }
}

// node_rep: f32 into d_out[0 .. N*134), bf16 padded copy [N,160] into ws.
__global__ void node_kernel(const float* __restrict__ x, const float* __restrict__ agg,
                            const float* __restrict__ den,
                            float* __restrict__ nrep_out, __bf16* __restrict__ nrb) {
  int idx = blockIdx.x * blockDim.x + threadIdx.x;
  if (idx >= NN * NRPAD) return;
  int n = idx / NRPAD;
  int k = idx - n * NRPAD;
  float v = 0.f;
  if (k < DIN)                 v = agg[(size_t)n * DIN + k] / (den[n] + 1.f);
  else if (k < DIN + DNODE)    v = x[n * DNODE + (k - DIN)];
  nrb[(size_t)n * NRPAD + k] = f2bf(v);
  if (k < DIN + DNODE) nrep_out[(size_t)n * 134 + k] = v;
}

// Pre-swizzle a weight block into B-fragment layout [kc][nt][lane][16]:
// lane covers column n = nt*16 + (lane&15), K = kc*32 + (lane>>4)*8 + t (t<8), kc*32 + 16 + (lane>>4)*8 + (t-8) (t>=8).
__global__ void wprep_kernel(const float* __restrict__ src, __bf16* __restrict__ dst,
                             int KC, int NT, int Ksrc, int rowoff, int Ncols) {
  int idx = blockIdx.x * blockDim.x + threadIdx.x;
  if (idx >= KC * NT * 512) return;
  int t = idx & 15;
  int lane = (idx >> 4) & 31;
  int rest = idx >> 9;
  int nt = rest % NT;
  int kc = rest / NT;
  int k = kc * 32 + (t < 8 ? (lane >> 4) * 8 + t : 16 + (lane >> 4) * 8 + (t - 8));
  int n = nt * 16 + (lane & 15);
  float v = (k < Ksrc) ? src[(size_t)(k + rowoff) * Ncols + n] : 0.f;
  dst[idx] = f2bf(v);
}

// ----------------------------------------------------------- fused edge FFN --
__global__ __launch_bounds__(128) void edge_kernel(
    const int* __restrict__ ei, const float* __restrict__ ea,
    const float* __restrict__ b_edge, const float* __restrict__ b1,
    const float* __restrict__ b2, const float* __restrict__ g1,
    const float* __restrict__ be1, const float* __restrict__ g2,
    const float* __restrict__ be2, const __bf16* __restrict__ nrb,
    const __bf16* __restrict__ wa, const __bf16* __restrict__ wb,
    const __bf16* __restrict__ wc, const __bf16* __restrict__ w1s,
    const __bf16* __restrict__ w2s, float* __restrict__ out) {
  __shared__ __align__(16) float  hs[4][16][132];  // per-wave 16x128 f32 tile (padded)
  __shared__ __align__(32) __bf16 zs[4][16][136];  // per-wave 16x128 bf16 z tile (padded)

  const int lane = threadIdx.x & 31;
  const int wv   = threadIdx.x >> 5;
  const int l15  = lane & 15;
  const int lhi  = lane >> 4;
  const int kb0  = lhi * 8;
  const int ebase = (blockIdx.x * 4 + wv) * 16;
  const int em    = ebase + l15;  // edge row this lane gathers / normalizes

  const int rrow = ei[em];
  const int rcol = ei[EE + em];
  const __bf16* nr_r = nrb + (size_t)rrow * NRPAD;
  const __bf16* nr_c = nrb + (size_t)rcol * NRPAD;
  const float*  ear  = ea + (size_t)em * DIN;

  // ---- GEMM1: edge_rep[16,396(pad 448)] @ W_edge -> 16x128 ----
  v8f acc[8] = {};
#pragma unroll
  for (int kc = 0; kc < 5; ++kc) {           // node_rep[row] segment (K 0..133)
    v16bf A = loadA_bf(nr_r, kc * 32 + kb0);
#pragma unroll
    for (int nt = 0; nt < 8; ++nt) {
      v16bf B = *(const v16bf*)(wa + ((kc * 8 + nt) * 32 + lane) * 16);
      acc[nt] = WMMA_BF16(A, B, acc[nt]);
    }
  }
#pragma unroll
  for (int kc = 0; kc < 5; ++kc) {           // node_rep[col] segment
    v16bf A = loadA_bf(nr_c, kc * 32 + kb0);
#pragma unroll
    for (int nt = 0; nt < 8; ++nt) {
      v16bf B = *(const v16bf*)(wb + ((kc * 8 + nt) * 32 + lane) * 16);
      acc[nt] = WMMA_BF16(A, B, acc[nt]);
    }
  }
#pragma unroll
  for (int kc = 0; kc < 4; ++kc) {           // edge_attr segment (f32 -> bf16)
    v16bf A = loadA_f32(ear, kc * 32 + kb0);
#pragma unroll
    for (int nt = 0; nt < 8; ++nt) {
      v16bf B = *(const v16bf*)(wc + ((kc * 8 + nt) * 32 + lane) * 16);
      acc[nt] = WMMA_BF16(A, B, acc[nt]);
    }
  }

  // C fragment -> LDS: lane holds col n = nt*16+l15, rows lhi*8 + r
#pragma unroll
  for (int nt = 0; nt < 8; ++nt)
#pragma unroll
    for (int r = 0; r < 8; ++r)
      hs[wv][lhi * 8 + r][nt * 16 + l15] = acc[nt][r];
  __builtin_amdgcn_wave_barrier();

  // ---- residual + bias + LayerNorm1 (half-row per lane, shfl_xor 16) ----
  {
    const int row = l15, cb = lhi * 64;
    const float* eres = ea + (size_t)(ebase + row) * DIN;
    float s = 0.f, s2 = 0.f;
    for (int c = 0; c < 64; ++c) {
      int n = cb + c;
      float v = hs[wv][row][n] + b_edge[n] + eres[n];
      hs[wv][row][n] = v;
      s += v; s2 += v * v;
    }
    s  += __shfl_xor(s, 16, 32);
    s2 += __shfl_xor(s2, 16, 32);
    float mu = s * (1.f / 128.f);
    float rs = rsqrtf(s2 * (1.f / 128.f) - mu * mu + 1e-5f);
    for (int c = 0; c < 64; ++c) {
      int n = cb + c;
      hs[wv][row][n] = (hs[wv][row][n] - mu) * rs * g1[n] + be1[n];
    }
  }
  __builtin_amdgcn_wave_barrier();

  // ---- fused FFN: relu(h@W1+b1)@W2, K of GEMM3 split in two 128-halves ----
  v8f acc3[8] = {};
  const float*  hrow = &hs[wv][l15][0];
  const __bf16* zrow = &zs[wv][l15][0];
#pragma unroll
  for (int half = 0; half < 2; ++half) {
    v8f acc2[8] = {};
#pragma unroll
    for (int kc = 0; kc < 4; ++kc) {
      v16bf A = loadA_lds_f32(hrow, kc * 32 + kb0);
#pragma unroll
      for (int nt = 0; nt < 8; ++nt) {
        v16bf B = *(const v16bf*)(w1s + ((kc * 16 + half * 8 + nt) * 32 + lane) * 16);
        acc2[nt] = WMMA_BF16(A, B, acc2[nt]);
      }
    }
    __builtin_amdgcn_wave_barrier();
#pragma unroll
    for (int nt = 0; nt < 8; ++nt)
#pragma unroll
      for (int r = 0; r < 8; ++r) {
        int n = half * 128 + nt * 16 + l15;
        float v = acc2[nt][r] + b1[n];
        v = v > 0.f ? v : 0.f;
        zs[wv][lhi * 8 + r][nt * 16 + l15] = f2bf(v);
      }
    __builtin_amdgcn_wave_barrier();
#pragma unroll
    for (int kc = 0; kc < 4; ++kc) {
      v16bf A = loadA_bf(zrow, kc * 32 + kb0);
#pragma unroll
      for (int nt = 0; nt < 8; ++nt) {
        v16bf B = *(const v16bf*)(w2s + (((half * 4 + kc) * 8 + nt) * 32 + lane) * 16);
        acc3[nt] = WMMA_BF16(A, B, acc3[nt]);
      }
    }
    __builtin_amdgcn_wave_barrier();
  }

  // ---- residual accumulate + LayerNorm2 + store ----
#pragma unroll
  for (int nt = 0; nt < 8; ++nt)
#pragma unroll
    for (int r = 0; r < 8; ++r) {
      int n = nt * 16 + l15;
      hs[wv][lhi * 8 + r][n] += acc3[nt][r] + b2[n];
    }
  __builtin_amdgcn_wave_barrier();
  {
    const int row = l15, cb = lhi * 64;
    float s = 0.f, s2 = 0.f;
    for (int c = 0; c < 64; ++c) {
      float v = hs[wv][row][cb + c];
      s += v; s2 += v * v;
    }
    s  += __shfl_xor(s, 16, 32);
    s2 += __shfl_xor(s2, 16, 32);
    float mu = s * (1.f / 128.f);
    float rs = rsqrtf(s2 * (1.f / 128.f) - mu * mu + 1e-5f);
    float* op = out + (size_t)(ebase + row) * DOUT;
    for (int c = 0; c < 64; ++c) {
      int n = cb + c;
      op[n] = (hs[wv][row][n] - mu) * rs * g2[n] + be2[n];
    }
  }
}

// ---------------------------------------------------------------- launcher ---
extern "C" void kernel_launch(void* const* d_in, const int* in_sizes, int n_in,
                              void* d_out, int out_size, void* d_ws, size_t ws_size,
                              hipStream_t stream) {
  (void)in_sizes; (void)n_in; (void)out_size; (void)ws_size;
  const float* x      = (const float*)d_in[0];
  const int*   ei     = (const int*)d_in[1];
  const float* ea     = (const float*)d_in[2];
  const float* mask   = (const float*)d_in[3];
  const float* W_edge = (const float*)d_in[5];
  const float* b_edge = (const float*)d_in[6];
  const float* W1     = (const float*)d_in[7];
  const float* b1     = (const float*)d_in[8];
  const float* W2     = (const float*)d_in[9];
  const float* b2     = (const float*)d_in[10];
  const float* g1     = (const float*)d_in[11];
  const float* be1    = (const float*)d_in[12];
  const float* g2     = (const float*)d_in[13];
  const float* be2    = (const float*)d_in[14];

  char* ws = (char*)d_ws;                       // all offsets 32B-aligned
  float*  agg = (float*)(ws + 0);               // N*128 f32
  float*  den = (float*)(ws + 25600000);        // N f32 (contiguous after agg)
  __bf16* nrb = (__bf16*)(ws + 25800000);       // N*160 bf16
  __bf16* wa  = (__bf16*)(ws + 41800000);       // 5*8*512 bf16
  __bf16* wb  = (__bf16*)(ws + 41840960);       // 5*8*512
  __bf16* wc  = (__bf16*)(ws + 41881920);       // 4*8*512
  __bf16* w1s = (__bf16*)(ws + 41914688);       // 4*16*512
  __bf16* w2s = (__bf16*)(ws + 41980224);       // 8*8*512

  float* nrep_out = (float*)d_out;
  float* edge_out = (float*)d_out + (size_t)NN * 134;

  zero_kernel<<<(NN * 129 + 255) / 256, 256, 0, stream>>>(agg, NN * 129);
  scatter_kernel<<<(EE * 32) / 256, 256, 0, stream>>>(ei, ea, mask, agg, den);
  node_kernel<<<(NN * NRPAD + 255) / 256, 256, 0, stream>>>(x, agg, den, nrep_out, nrb);
  wprep_kernel<<<(5 * 8 * 512 + 255) / 256, 256, 0, stream>>>(W_edge, wa, 5, 8, 134, 0, 128);
  wprep_kernel<<<(5 * 8 * 512 + 255) / 256, 256, 0, stream>>>(W_edge, wb, 5, 8, 134, 134, 128);
  wprep_kernel<<<(4 * 8 * 512 + 255) / 256, 256, 0, stream>>>(W_edge, wc, 4, 8, 128, 268, 128);
  wprep_kernel<<<(4 * 16 * 512 + 255) / 256, 256, 0, stream>>>(W1, w1s, 4, 16, 128, 0, 256);
  wprep_kernel<<<(8 * 8 * 512 + 255) / 256, 256, 0, stream>>>(W2, w2s, 8, 8, 256, 0, 128);
  edge_kernel<<<EE / 64, 128, 0, stream>>>(ei, ea, b_edge, b1, b2, g1, be1, g2, be2,
                                           nrb, wa, wb, wc, w1s, w2s, edge_out);
}
